// GCNForDialog_18923625906416
// MI455X (gfx1250) — compile-verified, matrix operand order checked by reading, not run on hardware
//
#include <hip/hip_runtime.h>
#include <hip/hip_bf16.h>
#include <cstdint>

// ---------------------------------------------------------------------------
// GCNForDialog on MI455X (gfx1250):
//   Double-buffered TDM (tensor_load_to_lds) tile staging overlapped with
//   bf16 WMMA GEMMs + L2-resident f32 scatter-sum aggregation.
// ---------------------------------------------------------------------------

#define N_NODES   20000
#define N_EDGES   320000
#define D_IN      512
#define D_H       256
#define N_CLASSES 4
#define N_LAYERS  64

typedef __attribute__((ext_vector_type(16))) __bf16   v16bf;
typedef __attribute__((ext_vector_type(8)))  __bf16   v8bf;
typedef __attribute__((ext_vector_type(8)))  float    v8f;
typedef __attribute__((ext_vector_type(4)))  unsigned u32x4;
typedef __attribute__((ext_vector_type(8)))  unsigned u32x8;

// ---------------------------------------------------------------------------
// TDM 2D tile load: builds D# group0 (128b) + group1 (256b) per CDNA5 ISA §8
// and issues tensor_load_to_lds with groups 2/3 omitted (NULL => <=2D tensor).
// Inline asm keeps this portable across both toolchains' builtin arities.
// data_size = 2 bytes (bf16). Caller is responsible for TENSORcnt waits.
// ---------------------------------------------------------------------------
__device__ __forceinline__ void tdm_load_tile_2d_bf16(
    unsigned lds_off, const void* gaddr, unsigned tile_x, unsigned tile_y,
    unsigned tensor_x, unsigned tensor_y, unsigned stride_x) {
  const unsigned long long ga = (unsigned long long)(uintptr_t)gaddr;
  u32x4 g0;
  g0[0] = 1u;                                   // count=1, user mode, no gather
  g0[1] = lds_off;                              // lds_addr
  g0[2] = (unsigned)ga;                         // global_addr[31:0]
  g0[3] = (unsigned)((ga >> 32) & 0x01FFFFFFull) | (2u << 30);  // [56:32]|type=2
  u32x8 g1;
  g1[0] = 1u << 16;                             // workgroup_mask=0, data_size=1(2B)
  g1[1] = (tensor_x & 0xFFFFu) << 16;           // tensor_dim0[15:0] @ bits 63:48
  g1[2] = ((tensor_x >> 16) & 0xFFFFu) | ((tensor_y & 0xFFFFu) << 16);
  g1[3] = ((tensor_y >> 16) & 0xFFFFu) | (tile_x << 16);  // tile_dim0
  g1[4] = tile_y & 0xFFFFu;                     // tile_dim1 (tile_dim2=0)
  g1[5] = stride_x;                             // tensor_dim0_stride[31:0]
  g1[6] = 0u;                                   // stride hi / dim1_stride lo
  g1[7] = 0u;
  asm volatile("tensor_load_to_lds %0, %1" ::"s"(g0), "s"(g1) : "memory");
}

// ---------------------------------------------------------------------------
// fp32 -> bf16 (row-major copy): used for x
// ---------------------------------------------------------------------------
__global__ void f32_to_bf16_kernel(const float* __restrict__ src,
                                   __bf16* __restrict__ dst, int n) {
  int i = blockIdx.x * blockDim.x + threadIdx.x;
  if (i < n) dst[i] = (__bf16)src[i];
}

// ---------------------------------------------------------------------------
// fp32 [K][Dout] -> bf16 [Dout][K] transpose-convert (writes coalesced).
// ---------------------------------------------------------------------------
__global__ void w0_transpose_kernel(const float* __restrict__ src,
                                    __bf16* __restrict__ dst) {
  int i = blockIdx.x * blockDim.x + threadIdx.x;     // i = c*D_IN + k
  if (i >= D_IN * D_H) return;
  int c = i / D_IN, k = i % D_IN;
  dst[i] = (__bf16)src[(size_t)k * D_H + c];
}

__global__ void ws_transpose_kernel(const float* __restrict__ src,
                                    __bf16* __restrict__ dst) {
  int i = blockIdx.x * blockDim.x + threadIdx.x;     // layer-major
  if (i >= (N_LAYERS - 1) * D_H * D_H) return;
  int l = i >> 16;                                   // 65536 = D_H*D_H
  int r = i & 0xFFFF;
  int c = r >> 8, k = r & 0xFF;
  dst[i] = (__bf16)src[((size_t)l << 16) + (size_t)k * D_H + c];
}

// ---------------------------------------------------------------------------
// Degree / normalization
// ---------------------------------------------------------------------------
__global__ void deg_init_kernel(float* __restrict__ deg, int n) {
  int i = blockIdx.x * blockDim.x + threadIdx.x;
  if (i < n) deg[i] = 1.0f;  // self loop
}

__global__ void deg_accum_kernel(const long long* __restrict__ col,
                                 float* __restrict__ deg, int e) {
  int i = blockIdx.x * blockDim.x + threadIdx.x;
  if (i < e) atomicAdd(&deg[(int)col[i]], 1.0f);
}

__global__ void dinv_kernel(float* __restrict__ deg, int n) {
  int i = blockIdx.x * blockDim.x + threadIdx.x;
  if (i < n) {
    float d = deg[i];
    deg[i] = (d > 0.0f) ? rsqrtf(d) : 0.0f;
  }
}

__global__ void norm_kernel(const long long* __restrict__ row,
                            const long long* __restrict__ col,
                            const float* __restrict__ dinv,
                            float* __restrict__ norm, int e) {
  int i = blockIdx.x * blockDim.x + threadIdx.x;
  if (i < e) norm[i] = dinv[(int)row[i]] * dinv[(int)col[i]];
}

// ---------------------------------------------------------------------------
// GEMM: C[N x 256] = A[N x K] (bf16) @ W[K x 256], W pre-transposed to
// Wt[256][K] bf16. Block = 128 threads = 4 waves covering 16 rows x 256 cols.
//
// Software-pipelined with double-buffered LDS: wave 0 TDM-loads tile s+1 into
// buf[nxt] (no wait), all waves run 4 WMMAs per wave on buf[cur] (one A
// fragment reused across the wave's 4 column tiles), then barrier +
// s_wait_tensorcnt 0 + barrier publishes the next tile. TDM runs fully under
// the matrix math. Loop unrolled by 2 so the buffer parity (and therefore the
// ds_load_b128 offsets) are compile-time immediates.
//
// Fragment layout per ISA 16-bit A-matrix map: lane<16 holds K 0..7 / 16..23,
// lane>=16 holds K 8..15 / 24..31; B symmetric with lane = column.
// ---------------------------------------------------------------------------
template <int KDIM>
__global__ __launch_bounds__(128) void gemm_tdm_wmma_kernel(
    const __bf16* __restrict__ A, const __bf16* __restrict__ Wt,
    float* __restrict__ C) {
  __shared__ __bf16 As[2][16][32];    // 2 KB
  __shared__ __bf16 Bs[2][D_H][32];   // 32 KB
  const int tid  = threadIdx.x;
  const int lane = tid & 31;
  const int wave = tid >> 5;
  const int half = lane >> 4;
  const int sub  = lane & 15;
  const int rowBase = blockIdx.x * 16;
  constexpr int STEPS = KDIM / 32;

  const unsigned ldsA0 = (unsigned)(uintptr_t)(&As[0][0][0]);
  const unsigned ldsA1 = (unsigned)(uintptr_t)(&As[1][0][0]);
  const unsigned ldsB0 = (unsigned)(uintptr_t)(&Bs[0][0][0]);
  const unsigned ldsB1 = (unsigned)(uintptr_t)(&Bs[1][0][0]);

  v8f acc[4] = {v8f{}, v8f{}, v8f{}, v8f{}};

  // Prologue: stage tile 0 into buffer 0
  if (wave == 0) {
    tdm_load_tile_2d_bf16(ldsA0, A + (size_t)rowBase * KDIM,
                          32, 16, KDIM, 16, KDIM);
    tdm_load_tile_2d_bf16(ldsB0, Wt, 32, D_H, KDIM, D_H, KDIM);
    __builtin_amdgcn_s_wait_tensorcnt(0);
  }
  __syncthreads();

#pragma unroll 2
  for (int s = 0; s < STEPS; ++s) {
    const int cur = s & 1;
    const int k1 = (s + 1) * 32;

    // Kick off next tile's DMA into the other buffer (no wait here: the
    // tensor DMA overlaps the WMMAs below).
    if (wave == 0 && k1 < KDIM) {
      tdm_load_tile_2d_bf16(cur ? ldsA0 : ldsA1,
                            A + (size_t)rowBase * KDIM + k1,
                            32, 16, KDIM, 16, KDIM);
      tdm_load_tile_2d_bf16(cur ? ldsB0 : ldsB1, Wt + k1,
                            32, D_H, KDIM, D_H, KDIM);
    }

    // Compute on the current buffer
    const v8bf alo = *(const v8bf*)(&As[cur][sub][half * 8]);
    const v8bf ahi = *(const v8bf*)(&As[cur][sub][16 + half * 8]);
    const v16bf a = __builtin_shufflevector(alo, ahi, 0, 1, 2, 3, 4, 5, 6, 7,
                                            8, 9, 10, 11, 12, 13, 14, 15);
#pragma unroll
    for (int t = 0; t < 4; ++t) {
      const int col = wave * 64 + t * 16 + sub;
      const v8bf blo = *(const v8bf*)(&Bs[cur][col][half * 8]);
      const v8bf bhi = *(const v8bf*)(&Bs[cur][col][16 + half * 8]);
      const v16bf b = __builtin_shufflevector(blo, bhi, 0, 1, 2, 3, 4, 5, 6, 7,
                                              8, 9, 10, 11, 12, 13, 14, 15);
      acc[t] = __builtin_amdgcn_wmma_f32_16x16x32_bf16(
          false, a, false, b, (short)0, acc[t], false, false);
    }

    __syncthreads();                               // done reading buf[cur]
    if (wave == 0) __builtin_amdgcn_s_wait_tensorcnt(0);  // tile s+1 landed
    __syncthreads();                               // buf[nxt] visible to all
  }

  // C/D layout: VGPR r -> (M = half*8 + r, N = sub)
#pragma unroll
  for (int t = 0; t < 4; ++t) {
#pragma unroll
    for (int r = 0; r < 8; ++r) {
      C[(size_t)(rowBase + half * 8 + r) * D_H + (wave * 64 + t * 16 + sub)] =
          acc[t][r];
    }
  }
}

// ---------------------------------------------------------------------------
// Self-loop contribution: agg[i][d] = dinv[i]^2 * T[i][d] (zero-inits agg)
// ---------------------------------------------------------------------------
__global__ void selfloop_kernel(const float* __restrict__ dinv,
                                const float* __restrict__ T,
                                float* __restrict__ agg, int total) {
  int i = blockIdx.x * blockDim.x + threadIdx.x;
  if (i >= total) return;
  float dv = dinv[i / D_H];
  agg[i] = dv * dv * T[i];
}

// ---------------------------------------------------------------------------
// Edge scatter: agg[col[e]] += norm[e] * T[row[e]]  (float4 gather, f32
// atomics resolving in the 192MB L2 — all live tensors are L2 resident)
// ---------------------------------------------------------------------------
__global__ void edge_scatter_kernel(const long long* __restrict__ row,
                                    const long long* __restrict__ col,
                                    const float* __restrict__ norm,
                                    const float* __restrict__ T,
                                    float* __restrict__ agg, int e) {
  const int chunks = D_H / 4;  // 64
  int gid = blockIdx.x * blockDim.x + threadIdx.x;
  if (gid >= e * chunks) return;
  int ei = gid / chunks;
  int c  = (gid % chunks) * 4;
  int r  = (int)row[ei];
  int cl = (int)col[ei];
  float nv = norm[ei];
  const float4 v = *(const float4*)(T + (size_t)r * D_H + c);
  float* dst = agg + (size_t)cl * D_H + c;
  atomicAdd(dst + 0, v.x * nv);
  atomicAdd(dst + 1, v.y * nv);
  atomicAdd(dst + 2, v.z * nv);
  atomicAdd(dst + 3, v.w * nv);
}

// ---------------------------------------------------------------------------
// Bias + ReLU, emitting bf16 activations for the next layer's WMMA GEMM
// ---------------------------------------------------------------------------
__global__ void bias_relu_bf16_kernel(const float* __restrict__ agg,
                                      const float* __restrict__ bias,
                                      __bf16* __restrict__ hb, int total) {
  int i = blockIdx.x * blockDim.x + threadIdx.x;
  if (i >= total) return;
  float v = agg[i] + bias[i & (D_H - 1)];
  hb[i] = (__bf16)(v > 0.0f ? v : 0.0f);
}

// ---------------------------------------------------------------------------
// Classifier: out[n] = h[n] @ Wc + bc   (256 -> 4, tiny, plain VALU)
// ---------------------------------------------------------------------------
__global__ void classifier_kernel(const __bf16* __restrict__ hb,
                                  const float* __restrict__ Wc,
                                  const float* __restrict__ bc,
                                  float* __restrict__ out, int n) {
  int i = blockIdx.x * blockDim.x + threadIdx.x;
  if (i >= n) return;
  float a0 = bc[0], a1 = bc[1], a2 = bc[2], a3 = bc[3];
  const __bf16* hr = hb + (size_t)i * D_H;
#pragma unroll 4
  for (int d = 0; d < D_H; ++d) {
    float hv = (float)hr[d];
    a0 += hv * Wc[d * 4 + 0];
    a1 += hv * Wc[d * 4 + 1];
    a2 += hv * Wc[d * 4 + 2];
    a3 += hv * Wc[d * 4 + 3];
  }
  out[i * 4 + 0] = a0;
  out[i * 4 + 1] = a1;
  out[i * 4 + 2] = a2;
  out[i * 4 + 3] = a3;
}

// ---------------------------------------------------------------------------
// Host-side orchestration (graph-capture safe: launches only, deterministic)
// ---------------------------------------------------------------------------
static inline int cdiv(int a, int b) { return (a + b - 1) / b; }

extern "C" void kernel_launch(void* const* d_in, const int* in_sizes, int n_in,
                              void* d_out, int out_size, void* d_ws,
                              size_t ws_size, hipStream_t stream) {
  (void)in_sizes; (void)n_in; (void)out_size; (void)ws_size;

  const float*     x   = (const float*)d_in[0];
  const long long* ei  = (const long long*)d_in[1];   // int64 per reference
  const float*     W0  = (const float*)d_in[2];
  const float*     b0  = (const float*)d_in[3];
  const float*     Ws  = (const float*)d_in[4];
  const float*     bs  = (const float*)d_in[5];
  const float*     Wc  = (const float*)d_in[6];
  const float*     bc  = (const float*)d_in[7];
  float*           out = (float*)d_out;

  const long long* rowIdx = ei;             // edge_index[0] = source
  const long long* colIdx = ei + N_EDGES;   // edge_index[1] = target

  // Workspace carve-up (256B aligned slices)
  char* ws = (char*)d_ws;
  size_t off = 0;
  auto carve = [&](size_t bytes) -> void* {
    off = (off + 255) & ~((size_t)255);
    void* p = ws + off;
    off += bytes;
    return p;
  };
  float*  dinv = (float*)carve(sizeof(float) * N_NODES);
  float*  norm = (float*)carve(sizeof(float) * N_EDGES);
  __bf16* xb   = (__bf16*)carve(sizeof(__bf16) * (size_t)N_NODES * D_IN);
  __bf16* Wt0  = (__bf16*)carve(sizeof(__bf16) * (size_t)D_IN * D_H);
  __bf16* Wts  = (__bf16*)carve(sizeof(__bf16) * (size_t)(N_LAYERS - 1) *
                                D_H * D_H);
  float*  T    = (float*)carve(sizeof(float) * (size_t)N_NODES * D_H);
  float*  agg  = (float*)carve(sizeof(float) * (size_t)N_NODES * D_H);
  __bf16* hb   = (__bf16*)carve(sizeof(__bf16) * (size_t)N_NODES * D_H);

  const int total = N_NODES * D_H;

  // 1) One-shot conversions: x -> bf16, weights -> bf16 transposed [Dout][K]
  {
    int nx = N_NODES * D_IN;
    f32_to_bf16_kernel<<<cdiv(nx, 256), 256, 0, stream>>>(x, xb, nx);
    int n0 = D_IN * D_H;
    w0_transpose_kernel<<<cdiv(n0, 256), 256, 0, stream>>>(W0, Wt0);
    int n1 = (N_LAYERS - 1) * D_H * D_H;
    ws_transpose_kernel<<<cdiv(n1, 256), 256, 0, stream>>>(Ws, Wts);
  }

  // 2) Degrees and symmetric normalization coefficients
  deg_init_kernel<<<cdiv(N_NODES, 256), 256, 0, stream>>>(dinv, N_NODES);
  deg_accum_kernel<<<cdiv(N_EDGES, 256), 256, 0, stream>>>(colIdx, dinv,
                                                           N_EDGES);
  dinv_kernel<<<cdiv(N_NODES, 256), 256, 0, stream>>>(dinv, N_NODES);
  norm_kernel<<<cdiv(N_EDGES, 256), 256, 0, stream>>>(rowIdx, colIdx, dinv,
                                                      norm, N_EDGES);

  // 3) GCN layers
  const int gemmBlocks = N_NODES / 16;  // 1250
  const int scatterThreads = N_EDGES * (D_H / 4);
  for (int l = 0; l < N_LAYERS; ++l) {
    if (l == 0) {
      gemm_tdm_wmma_kernel<D_IN><<<gemmBlocks, 128, 0, stream>>>(xb, Wt0, T);
    } else {
      gemm_tdm_wmma_kernel<D_H><<<gemmBlocks, 128, 0, stream>>>(
          hb, Wts + (size_t)(l - 1) * D_H * D_H, T);
    }
    selfloop_kernel<<<cdiv(total, 256), 256, 0, stream>>>(dinv, T, agg, total);
    edge_scatter_kernel<<<cdiv(scatterThreads, 256), 256, 0, stream>>>(
        rowIdx, colIdx, norm, T, agg, N_EDGES);
    const float* bias = (l == 0) ? b0 : bs + (size_t)(l - 1) * D_H;
    bias_relu_bf16_kernel<<<cdiv(total, 256), 256, 0, stream>>>(agg, bias, hb,
                                                                total);
  }

  // 4) Classifier head
  classifier_kernel<<<cdiv(N_NODES, 256), 256, 0, stream>>>(hb, Wc, bc, out,
                                                            N_NODES);
}
